// PtrDecoder_55585466745318
// MI455X (gfx1250) — matrix operand order repaired and checked by reading, not compiled, not run on hardware
//
#include <hip/hip_runtime.h>
#include <math.h>

// ---- problem constants ----
#define L_TOTAL   100000
#define OUT_DIM   100000
#define HID       256
#define LDSW      264                   // padded LDS row stride (ushorts): kills bank conflicts
#define MROWS     32                    // encoder rows per workgroup iteration
#define NTILES    (L_TOTAL / MROWS)     // 3125, exact
#define NWAVES    16                    // waves per workgroup in score kernel
// d_out layout: [out_full 100000][h_new 256][c_new 256][attn_weights 100000]
#define DO_H      100000
#define DO_C      100256
#define DO_ATTN   100512

typedef __attribute__((ext_vector_type(16))) __bf16          v16bf;
typedef __attribute__((ext_vector_type(8)))  float           v8f;
typedef __attribute__((ext_vector_type(8)))  unsigned short  ushort8;
typedef unsigned short ushort_t;

#define LOG2E 1.44269504088896340736f

__device__ __forceinline__ ushort_t f32_to_bf16(float f) {
  union { float f; unsigned u; } x; x.f = f;
  unsigned u = x.u;
  unsigned r = u + 0x7FFFu + ((u >> 16) & 1u);   // round-to-nearest-even
  return (ushort_t)(r >> 16);
}

// branchless tanh: native v_tanh_f32 when the builtin exists, else exp2+rcp
__device__ __forceinline__ float fast_tanh(float x) {
#if __has_builtin(__builtin_amdgcn_tanhf)
  return __builtin_amdgcn_tanhf(x);
#else
  float xc = fminf(fmaxf(x, -10.0f), 10.0f);           // overflow-safe, branchless
  float t  = __builtin_amdgcn_exp2f(xc * (2.0f * LOG2E));
  return (t - 1.0f) * __builtin_amdgcn_rcpf(t + 1.0f);
#endif
}

// branchless exp for x <= 0 (softmax: x - max)
__device__ __forceinline__ float fast_exp_nonpos(float x) {
  return __builtin_amdgcn_exp2f(x * LOG2E);
}

__device__ __forceinline__ float fast_log(float x) {
  return __builtin_amdgcn_logf(x) * 0.6931471805599453f;
}

__device__ __forceinline__ float sigmoidf_(float x) {
  return __builtin_amdgcn_rcpf(1.0f + __builtin_amdgcn_exp2f(-x * LOG2E));
}

// ===================================================================
// K1: LSTM cell + attention bias precompute + attn_w -> bf16 WMMA-A repack
// one workgroup of 1024 threads
// ===================================================================
__global__ __launch_bounds__(1024)
void lstm_prep_kernel(const int* __restrict__ tok,
                      const float* __restrict__ h0,
                      const float* __restrict__ c0,
                      const float* __restrict__ embedding,
                      const float* __restrict__ w_ih,
                      const float* __restrict__ w_hh,
                      const float* __restrict__ b_ih,
                      const float* __restrict__ b_hh,
                      const float* __restrict__ attn_w,
                      const float* __restrict__ attn_b,
                      float* __restrict__ d_out,
                      float* __restrict__ ws_bias,
                      ushort_t* __restrict__ ws_A) {
  __shared__ float sx[HID], sh[HID], sg[4 * HID], shn[HID];
  const int t = threadIdx.x;

  if (t < HID) {
    sx[t] = embedding[(size_t)tok[0] * HID + t];
    sh[t] = h0[t];
  }
  __syncthreads();

  // one gate per thread: 512 MACs
  {
    float g = b_ih[t] + b_hh[t];
    const float* wi = w_ih + (size_t)t * HID;
    const float* wh = w_hh + (size_t)t * HID;
    #pragma unroll 4
    for (int k = 0; k < HID; ++k) g += sx[k] * wi[k] + sh[k] * wh[k];
    sg[t] = g;
  }
  __syncthreads();

  if (t < HID) {  // torch gate order i,f,g,o
    float ig = sigmoidf_(sg[t]);
    float fg = sigmoidf_(sg[HID + t]);
    float gg = fast_tanh(sg[2 * HID + t]);
    float og = sigmoidf_(sg[3 * HID + t]);
    float cn = fg * c0[t] + ig * gg;
    float hn = og * fast_tanh(cn);
    shn[t] = hn;
    d_out[DO_H + t] = hn;
    d_out[DO_C + t] = cn;
  }
  __syncthreads();

  // bias[j] = attn_b[j] + attn_w[j, 256:512] . h_new   (constant over rows)
  if (t < HID) {
    float b = attn_b[t];
    const float* aw = attn_w + (size_t)t * (2 * HID) + HID;
    #pragma unroll 4
    for (int k = 0; k < HID; ++k) b += shn[k] * aw[k];
    ws_bias[t] = b;
  }

  // repack attn_w[:, 0:256] into WMMA bf16 **A** layout:
  // idx = (((T*8 + c)*32 + lane)*16 + e)
  // lane = m + 16*half ; j = T*16 + m ; element e <-> k = c*32 + 8*half + (e<8 ? e : e+8)
  for (int idx = t; idx < 16 * 8 * 32 * 16; idx += 1024) {
    int e    = idx & 15;
    int lane = (idx >> 4) & 31;
    int c    = (idx >> 9) & 7;
    int T    = idx >> 12;
    int m    = lane & 15;
    int half = lane >> 4;
    int j    = T * 16 + m;
    int k    = c * 32 + 8 * half + (e < 8 ? e : e + 8);
    ws_A[idx] = f32_to_bf16(attn_w[(size_t)j * (2 * HID) + k]);
  }
}

// ===================================================================
// K2: scores[row] = out_b + sum_j out_w[j] * tanh(enc[row,:] . attn_w[j,:256] + bias[j])
// Transposed WMMA: A = attn_w tile (j on M, resident 64 VGPRs),
//                  B = encoder tile transposed (rows on N, staged via LDS).
// 32 rows per iteration as TWO independent accumulator chains so the wave
// always has two independent WMMA streams in flight.
// C: VGPR r, lane = row + 16*half holds j = wave*16 + r + 8*half -> the
// j-reduction is an in-lane sum over 8 VGPRs + one shfl_xor(16).
// 512 persistent workgroups x 512 threads (16 waves).
// ===================================================================
__global__ __launch_bounds__(512)
void scores_kernel(const float* __restrict__ enc,
                   const ushort_t* __restrict__ Abuf,
                   const float* __restrict__ bias,
                   const float* __restrict__ out_w,
                   const float* __restrict__ out_b,
                   float* __restrict__ scores) {
  __shared__ ushort_t ldsA[MROWS * LDSW];      // 32 rows x 256 cols bf16, padded
  __shared__ float    ldsPart[NWAVES * MROWS]; // per-wave row partials

  const int tid  = threadIdx.x;
  const int wave = tid >> 5;          // 0..15 -> j column tile
  const int lane = tid & 31;
  const int half = lane >> 4;
  const int ln   = lane & 15;

  // resident A (attn_w): 8 K-chunks of [16 x 32] bf16 (8 VGPRs each)
  v16bf Aw[8];
  #pragma unroll
  for (int c = 0; c < 8; ++c)
    Aw[c] = ((const v16bf*)Abuf)[(wave * 8 + c) * 32 + lane];

  // per-accumulator-VGPR constants: j = wave*16 + r + 8*half
  float bv[8], owv[8];
  #pragma unroll
  for (int r = 0; r < 8; ++r) {
    const int j = wave * 16 + 8 * half + r;
    bv[r]  = bias[j];
    owv[r] = out_w[j];
  }
  const float outb = out_b[0];

  // staging assignment: thread t stages row t/16 (0..31), 16 consecutive cols
  const int srow = tid >> 4;
  const int scol = (tid & 15) * 16;

  for (int tile = blockIdx.x; tile < NTILES; tile += gridDim.x) {
    const int rowBase = tile * MROWS;

    // prefetch next tile this block will process (512 threads x 64B = 32KB)
    {
      int nt = tile + gridDim.x;
      if (nt < NTILES)
        __builtin_prefetch(enc + (size_t)nt * MROWS * HID + (size_t)tid * 16, 0, 3);
    }

    // stage 32x256 f32 -> bf16 into padded LDS
    {
      const float* src = enc + (size_t)(rowBase + srow) * HID + scol;
      #pragma unroll
      for (int e = 0; e < 16; ++e)
        ldsA[srow * LDSW + scol + e] = f32_to_bf16(src[e]);
    }
    __syncthreads();

    union { v8f v; float f[8]; } acc0, acc1;
    #pragma unroll
    for (int r = 0; r < 8; ++r) { acc0.f[r] = bv[r]; acc1.f[r] = bv[r]; }

    #pragma unroll
    for (int c = 0; c < 8; ++c) {
      // B lane layout: n = ln (row), element e <-> k = c*32 + 8*half + (e<8 ? e : e+8)
      // -> two 16B runs per lane in this lane's padded LDS row.
      union { v16bf v; ushort8 h[2]; } b0, b1;
      const ushort_t* base0 = &ldsA[ln * LDSW + c * 32 + 8 * half];
      const ushort_t* base1 = base0 + 16 * LDSW;
      b0.h[0] = *(const ushort8*)base0;
      b0.h[1] = *(const ushort8*)(base0 + 16);
      b1.h[0] = *(const ushort8*)base1;
      b1.h[1] = *(const ushort8*)(base1 + 16);
      acc0.v = __builtin_amdgcn_wmma_f32_16x16x32_bf16(
          false, Aw[c], false, b0.v, (short)0, acc0.v, false, false);
      acc1.v = __builtin_amdgcn_wmma_f32_16x16x32_bf16(
          false, Aw[c], false, b1.v, (short)0, acc1.v, false, false);
    }

    // in-lane j-reduction over the 8 accumulator VGPRs, then merge halves
    float p0 = 0.0f, p1 = 0.0f;
    #pragma unroll
    for (int r = 0; r < 8; ++r) {
      p0 += owv[r] * fast_tanh(acc0.f[r]);
      p1 += owv[r] * fast_tanh(acc1.f[r]);
    }
    p0 += __shfl_xor(p0, 16);           // lanes l and l+16 hold same row
    p1 += __shfl_xor(p1, 16);
    if (lane < 16) {
      ldsPart[wave * MROWS + lane]      = p0;   // rows 0..15
      ldsPart[wave * MROWS + 16 + lane] = p1;   // rows 16..31
    }
    __syncthreads();

    if (tid < MROWS) {
      float s = outb;
      #pragma unroll
      for (int w = 0; w < NWAVES; ++w) s += ldsPart[w * MROWS + tid];
      scores[rowBase + tid] = s;
    }
    __syncthreads();
  }
}

// ===================================================================
// K3: log-softmax stats (max, log-sum-exp) over 100000 scores; 1 block.
// ===================================================================
__global__ __launch_bounds__(1024)
void stats_kernel(const float* __restrict__ scores, float* __restrict__ stats) {
  __shared__ float red[1024];
  const int t = threadIdx.x;

  float m = -INFINITY;
  for (int i = t; i < L_TOTAL; i += 1024) m = fmaxf(m, scores[i]);
  red[t] = m;
  __syncthreads();
  for (int s = 512; s > 0; s >>= 1) {
    if (t < s) red[t] = fmaxf(red[t], red[t + s]);
    __syncthreads();
  }
  const float maxv = red[0];
  __syncthreads();

  float sum = 0.0f;
  for (int i = t; i < L_TOTAL; i += 1024) sum += fast_exp_nonpos(scores[i] - maxv);
  red[t] = sum;
  __syncthreads();
  for (int s = 512; s > 0; s >>= 1) {
    if (t < s) red[t] += red[t + s];
    __syncthreads();
  }
  if (t == 0) { stats[0] = maxv; stats[1] = fast_log(red[0]); }
}

// ===================================================================
// K4: init out_full = -inf and write attn_weights = scores - max - logZ
// ===================================================================
__global__ void finalize_kernel(const float* __restrict__ scores,
                                const float* __restrict__ stats,
                                float* __restrict__ d_out) {
  const int i = blockIdx.x * blockDim.x + threadIdx.x;
  if (i < OUT_DIM)  d_out[i] = -INFINITY;
  if (i < L_TOTAL)  d_out[DO_ATTN + i] = scores[i] - stats[0] - stats[1];
}

// ===================================================================
// K5: scatter out_full[encoder_inputs[i]] = attn_weights[i] (after init)
// ===================================================================
__global__ void scatter_kernel(const int* __restrict__ enc_in,
                               float* __restrict__ d_out) {
  const int i = blockIdx.x * blockDim.x + threadIdx.x;
  if (i < L_TOTAL) d_out[enc_in[i]] = d_out[DO_ATTN + i];
}

// ===================================================================
extern "C" void kernel_launch(void* const* d_in, const int* in_sizes, int n_in,
                              void* d_out, int out_size, void* d_ws, size_t ws_size,
                              hipStream_t stream) {
  const int*   tok        = (const int*)  d_in[0];
  const float* h0         = (const float*)d_in[1];
  const float* c0         = (const float*)d_in[2];
  const float* enc_out    = (const float*)d_in[3];
  const int*   enc_in     = (const int*)  d_in[4];
  const float* embedding  = (const float*)d_in[5];
  const float* w_ih       = (const float*)d_in[6];
  const float* w_hh       = (const float*)d_in[7];
  const float* b_ih       = (const float*)d_in[8];
  const float* b_hh       = (const float*)d_in[9];
  const float* attn_w     = (const float*)d_in[10];
  const float* attn_b     = (const float*)d_in[11];
  const float* out_w      = (const float*)d_in[12];
  const float* out_b      = (const float*)d_in[13];
  float* out = (float*)d_out;

  // workspace layout (floats): [bias 256][stats 2, pad to 512][scores 100000][A bf16 128KB]
  float*    ws_f      = (float*)d_ws;
  float*    ws_bias   = ws_f;
  float*    ws_stats  = ws_f + 256;
  float*    ws_scores = ws_f + 512;
  ushort_t* ws_A      = (ushort_t*)(ws_f + 512 + L_TOTAL);   // 32B-aligned offset

  lstm_prep_kernel<<<1, 1024, 0, stream>>>(tok, h0, c0, embedding, w_ih, w_hh,
                                           b_ih, b_hh, attn_w, attn_b,
                                           out, ws_bias, ws_A);

  scores_kernel<<<512, 512, 0, stream>>>(enc_out, ws_A, ws_bias, out_w, out_b,
                                         ws_scores);

  stats_kernel<<<1, 1024, 0, stream>>>(ws_scores, ws_stats);

  const int thr = 256;
  const int blk = (L_TOTAL + thr - 1) / thr;
  finalize_kernel<<<blk, thr, 0, stream>>>(ws_scores, ws_stats, out);
  scatter_kernel<<<blk, thr, 0, stream>>>(enc_in, out);
}
